// NCEObj_65893388255592
// MI455X (gfx1250) — matrix operand order repaired and checked by reading, not compiled, not run on hardware
//
#include <hip/hip_runtime.h>
#include <math.h>
#include <stdint.h>

// Problem constants (from reference): B=8, S=512, D=1024, N=B*S=4096
constexpr int Dd = 1024;
constexpr int Nn = 4096;
constexpr int SPLITS = 8;          // key-dimension splits (grid.y)
constexpr long long MASK_ID_LL = 8192LL;

// LDS staging for the A (embedding) block via TDM with LDS padding:
// pad 16B after every 1024B half-row -> half-row = 1040B, row = 2080B = 1040 bf16.
// Row-to-row bank rotation = 8 dwords (2-way worst-case conflict on b128 frags,
// the minimum achievable with TDM's 4-dword pad granularity + 16B b128 alignment).
constexpr int A_STRIDE = 1040;                     // bf16 elements per row in LDS
constexpr int LSE_LDS_BYTES = 64 * A_STRIDE * 2;   // 133120 B (reduction buf aliases)

typedef __attribute__((ext_vector_type(16))) __bf16 bf16x16;
typedef __attribute__((ext_vector_type(8)))  __bf16 bf16x8;
typedef __attribute__((ext_vector_type(8)))  float  f32x8;
typedef __attribute__((ext_vector_type(4)))  unsigned int uint32x4;
typedef __attribute__((ext_vector_type(8)))  int int32x8;
typedef __attribute__((ext_vector_type(4)))  int int32x4;

// ---------------------------------------------------------------------------
// Kernel 1: f32 -> bf16 conversion of both operands (one pass, streaming)
// ---------------------------------------------------------------------------
__global__ void cvt_kernel(const float* __restrict__ x, const float* __restrict__ e,
                           __bf16* __restrict__ xbf, __bf16* __restrict__ ebf, int n) {
  int i = blockIdx.x * blockDim.x + threadIdx.x;
  if (i < n) {
    xbf[i] = (__bf16)x[i];
    ebf[i] = (__bf16)e[i];
  }
}

// ---------------------------------------------------------------------------
// Kernel 2: numer[n] = dot(e_n, x_n) in full f32 (one wave32 per row)
// ---------------------------------------------------------------------------
__global__ void numer_kernel(const float* __restrict__ x, const float* __restrict__ e,
                             float* __restrict__ numer) {
  int wave = threadIdx.x >> 5, lane = threadIdx.x & 31;
  int row = blockIdx.x * 8 + wave;
  const float4* xr = (const float4*)(x + (size_t)row * Dd);
  const float4* er = (const float4*)(e + (size_t)row * Dd);
  float s = 0.f;
  for (int i = lane; i < Dd / 4; i += 32) {
    float4 a = er[i], b = xr[i];
    s += a.x * b.x + a.y * b.y + a.z * b.z + a.w * b.w;
  }
  #pragma unroll
  for (int off = 16; off > 0; off >>= 1) s += __shfl_down(s, off, 32);
  if (lane == 0) numer[row] = s;
}

// ---------------------------------------------------------------------------
// WMMA fragment loaders (CDNA5 16-bit layouts, wave32)
// A 16x32 (MxK): lane&15 = M row; lane-half picks K {0..7,16..23} / {8..15,24..31}
// B 32x16 (KxN): lane&15 = N col; lane-half picks K {0..7,16..23} / {8..15,24..31} (as A)
// ---------------------------------------------------------------------------
__device__ __forceinline__ bf16x16 load_a_lds(const __bf16* As, int mrow, int k0,
                                              int lane) {
  // TDM pad: +8 bf16 (16B) after each 1024B half-row; fragments never straddle it
  const __bf16* p = As + (size_t)(mrow + (lane & 15)) * A_STRIDE
                  + k0 + ((k0 >> 9) << 3) + ((lane >> 4) << 3);
  union { bf16x16 v; bf16x8 h[2]; } u;
  u.h[0] = *(const bf16x8*)(p);        // K = k0+koff .. +7      (ds_load_b128)
  u.h[1] = *(const bf16x8*)(p + 16);   // K = k0+16+koff .. +7   (ds_load_b128)
  return u.v;
}

__device__ __forceinline__ bf16x16 load_b(const __bf16* __restrict__ base, int col,
                                          int k0, int lane) {
  const __bf16* p = base + (size_t)(col + (lane & 15)) * Dd + k0 + ((lane >> 4) << 3);
  union { bf16x16 v; bf16x8 h[2]; } u;
  u.h[0] = *(const bf16x8*)(p);        // K = k0+koff .. +7
  u.h[1] = *(const bf16x8*)(p + 16);   // K = k0+16+koff .. +7
  return u.v;
}

// ---------------------------------------------------------------------------
// Kernel 3: fused GEMM + online logsumexp partials.
// Block = 64 query rows x 512 keys; 8 waves. The 64x1024 bf16 A block is staged
// once into LDS by the Tensor Data Mover (one descriptor, TENSORcnt-tracked),
// then all waves read A fragments from LDS; B streams from global. Scores never
// hit memory.
// ---------------------------------------------------------------------------
__global__ __launch_bounds__(256) void lse_kernel(const __bf16* __restrict__ ebf,
                                                  const __bf16* __restrict__ xbf,
                                                  float2* __restrict__ partial) {
  extern __shared__ __align__(16) char smem[];
  __bf16* As = (__bf16*)smem;                       // [64][A_STRIDE] bf16 (TDM-padded)
  const int wave = threadIdx.x >> 5;
  const int lane = threadIdx.x & 31;
  const int row0 = blockIdx.x * 64;
  const int split = blockIdx.y;

  // ---- Phase 0: TDM DMA of the A block (64 x 1024 bf16 = 128 KB + padding) ----
  if (wave == 0) {
    uint64_t gaddr = (uint64_t)(uintptr_t)(ebf + (size_t)row0 * Dd);
    uint32_t laddr = (uint32_t)(uintptr_t)As;       // LDS byte address = addr[31:0]

    uint32x4 g0;
    g0[0] = 1u;                                     // count=1, user descriptor
    g0[1] = laddr;                                  // lds_addr
    g0[2] = (uint32_t)gaddr;                        // global_addr[31:0]
    g0[3] = (uint32_t)((gaddr >> 32) & 0x01FFFFFFu) // global_addr[56:32]
          | 0x80000000u;                            // type=2 ("image")

    int32x8 g1;
    g1[0] = (1 << 16)                               // data_size = 2 bytes
          | (1 << 20)                               // pad_enable
          | (7 << 22)                               // pad_interval = 256 dwords (1024B)
          | (3 << 25);                              // pad_amount = 4 dwords (16B)
    g1[1] = (int)(1024u << 16);                     // tensor_dim0[15:0] = 1024
    g1[2] = (int)(64u << 16);                       // tensor_dim0[31:16]=0, tensor_dim1=64
    g1[3] = (int)(1024u << 16);                     // tensor_dim1[31:16]=0, tile_dim0=1024
    g1[4] = 64;                                     // tile_dim1=64, tile_dim2=0
    g1[5] = 1024;                                   // tensor_dim0_stride[31:0] = 1024
    g1[6] = 0;                                      // stride hi / tensor_dim1_stride
    g1[7] = 0;

    int32x4 z4 = {0, 0, 0, 0};                      // 2-D tensor: groups 2/3 unused
    int32x8 z8 = {0, 0, 0, 0, 0, 0, 0, 0};          // extra group (clang-23 6-arg form)
    __builtin_amdgcn_tensor_load_to_lds(g0, g1, z4, z4, z8, 0);
    __builtin_amdgcn_s_wait_tensorcnt(0);
  }
  __syncthreads();

  float mst[4][8], lst[4][8];
  #pragma unroll
  for (int m = 0; m < 4; ++m)
    #pragma unroll
    for (int r = 0; r < 8; ++r) { mst[m][r] = -INFINITY; lst[m][r] = 0.f; }

  const f32x8 vzero = {0.f, 0.f, 0.f, 0.f, 0.f, 0.f, 0.f, 0.f};

  #pragma unroll 1   // keep one copy: A ds_loads stay inside the k-loop body
  for (int g = 0; g < 4; ++g) {
    const int ncol0 = (split * 32 + g * 8 + wave) * 16;
    f32x8 acc[4];
    #pragma unroll
    for (int m = 0; m < 4; ++m) acc[m] = vzero;

    #pragma unroll 2 // partial unroll only: full unroll caused LICM->VGPR spill
    for (int k0 = 0; k0 < Dd; k0 += 32) {
      bf16x16 bfrag = load_b(xbf, ncol0, k0, lane);
      #pragma unroll
      for (int m = 0; m < 4; ++m) {
        bf16x16 afrag = load_a_lds(As, m * 16, k0, lane);
        acc[m] = __builtin_amdgcn_wmma_f32_16x16x32_bf16(
            /*neg_a=*/false, afrag, /*neg_b=*/false, bfrag,
            /*c_mod=*/(short)0, acc[m], /*reuse_a=*/false, /*reuse_b=*/false);
      }
    }

    // fold this N-tile's scores into running (max, sumexp) per lane slot
    #pragma unroll
    for (int m = 0; m < 4; ++m)
      #pragma unroll
      for (int r = 0; r < 8; ++r) {
        float s  = acc[m][r];
        float mo = mst[m][r];
        float mn = fmaxf(mo, s);
        lst[m][r] = lst[m][r] * __expf(mo - mn) + __expf(s - mn);
        mst[m][r] = mn;
      }
  }

  // A block is dead; alias the reduction buffer over it (after a barrier).
  __syncthreads();
  float2 (*red)[128] = (float2 (*)[128])smem;       // 64 x 128 float2 = 64 KB

  // scatter per-lane partials: row-local = m*16 + 8*(lane>=16) + r, col = wave*16 + (lane&15)
  const int rhalf = (lane >> 4) << 3;
  const int col = wave * 16 + (lane & 15);
  #pragma unroll
  for (int m = 0; m < 4; ++m)
    #pragma unroll
    for (int r = 0; r < 8; ++r)
      red[m * 16 + rhalf + r][col] = make_float2(mst[m][r], lst[m][r]);
  __syncthreads();

  // combine 128 partials per row; emit per-(row, split) partial (max, sumexp)
  if (threadIdx.x < 64) {
    float mm = -INFINITY, ll = 0.f;
    #pragma unroll 1
    for (int c = 0; c < 128; ++c) {
      float2 p = red[threadIdx.x][c];
      float mn = fmaxf(mm, p.x);
      ll = ll * __expf(mm - mn) + p.y * __expf(p.x - mn);
      mm = mn;
    }
    partial[(size_t)(row0 + threadIdx.x) * SPLITS + split] = make_float2(mm, ll);
  }
}

// ---------------------------------------------------------------------------
// Kernel 4: merge splits -> lse, apply mask + numer, reduce to scalar loss
// ---------------------------------------------------------------------------
__global__ void finalize_kernel(const float2* __restrict__ partial,
                                const float* __restrict__ numer,
                                const long long* __restrict__ tgt,
                                float* __restrict__ out) {
  __shared__ float ssum[256];
  __shared__ float scnt[256];
  float lsum = 0.f, lcnt = 0.f;
  for (int row = threadIdx.x; row < Nn; row += 256) {
    float mm = -INFINITY, ll = 0.f;
    #pragma unroll
    for (int s = 0; s < SPLITS; ++s) {
      float2 p = partial[(size_t)row * SPLITS + s];
      float mn = fmaxf(mm, p.x);
      ll = ll * __expf(mm - mn) + p.y * __expf(p.x - mn);
      mm = mn;
    }
    float lse = mm + __logf(ll);
    if (tgt[row] == MASK_ID_LL) { lsum += numer[row] - lse; lcnt += 1.f; }
  }
  ssum[threadIdx.x] = lsum;
  scnt[threadIdx.x] = lcnt;
  __syncthreads();
  for (int o = 128; o > 0; o >>= 1) {
    if (threadIdx.x < o) {
      ssum[threadIdx.x] += ssum[threadIdx.x + o];
      scnt[threadIdx.x] += scnt[threadIdx.x + o];
    }
    __syncthreads();
  }
  if (threadIdx.x == 0) out[0] = -(ssum[0] / scnt[0]);
}

// ---------------------------------------------------------------------------
extern "C" void kernel_launch(void* const* d_in, const int* in_sizes, int n_in,
                              void* d_out, int out_size, void* d_ws, size_t ws_size,
                              hipStream_t stream) {
  const float* x = (const float*)d_in[0];           // [B,S,D] f32
  const float* e = (const float*)d_in[1];           // [B,S,D] f32
  const long long* tgt = (const long long*)d_in[2]; // [B,S] int64
  float* out = (float*)d_out;

  // workspace layout: ebf(8MB) | xbf(8MB) | numer(16KB) | partial(256KB)
  char* w = (char*)d_ws;
  __bf16* ebf = (__bf16*)w;
  __bf16* xbf = ebf + (size_t)Nn * Dd;
  float* numer = (float*)(xbf + (size_t)Nn * Dd);
  float2* partial = (float2*)(numer + Nn);

  int n = Nn * Dd;
  cvt_kernel<<<(n + 255) / 256, 256, 0, stream>>>(x, e, xbf, ebf, n);
  numer_kernel<<<Nn / 8, 256, 0, stream>>>(x, e, numer);
  dim3 grid(Nn / 64, SPLITS);
  lse_kernel<<<grid, 256, LSE_LDS_BYTES, stream>>>(ebf, xbf, partial);
  finalize_kernel<<<1, 256, 0, stream>>>(partial, numer, tgt, out);
}
